// EEGEncoder_66812511257378
// MI455X (gfx1250) — matrix operand order repaired and checked by reading, not compiled, not run on hardware
//
#include <hip/hip_runtime.h>
#include <hip/hip_bf16.h>
#include <cmath>

// ---------------------------------------------------------------------------
// CDNA5 (gfx1250) wave32 WMMA types
// ---------------------------------------------------------------------------
typedef __bf16 v16bf __attribute__((ext_vector_type(16)));
typedef float  v8f   __attribute__((ext_vector_type(8)));

union BfFrag { v16bf v; unsigned int q[8]; };

__device__ __forceinline__ unsigned int f2bf_pk(float a, float b) {
    unsigned int ua = __builtin_bit_cast(unsigned int, a);
    ua += 0x7fffu + ((ua >> 16) & 1u);            // RNE
    unsigned int ub = __builtin_bit_cast(unsigned int, b);
    ub += 0x7fffu + ((ub >> 16) & 1u);
    return (ua >> 16) | (ub & 0xffff0000u);
}

// ---------------------------------------------------------------------------
// WMMA GEMM:  C[M,N] = act(A[M,K] @ B[K,N] + bias[N] (+ resid[M,N]))
// act: 0=none, 1=relu, 2=leaky(0.01)
// 512 threads = 16 waves (4x4); block tile 128x128; wave tile 32x32
// (2 A-frags x 2 B-frags -> 4 WMMAs per K-panel per wave).
// LDS holds bf16 pairs packed in u32; B tile stored transposed so both
// fragment types load with contiguous ds_load_b32.
// ---------------------------------------------------------------------------
#define GBM 128
#define GBN 128
#define GBK 32
#define GKK (GBK / 2)   // u32 per K-panel row

__global__ __launch_bounds__(512)
void wmma_gemm_kernel(const float* __restrict__ A, const float* __restrict__ Bm,
                      const float* __restrict__ bias, const float* __restrict__ resid,
                      float* __restrict__ C, int M, int N, int K, int act)
{
    __shared__ unsigned int lA[GBM][GKK + 1];   // [m][kk]  k = 2kk, 2kk+1
    __shared__ unsigned int lB[GBN][GKK + 1];   // [n][kk]  (transposed)

    const int tid  = threadIdx.x;
    const int lane = tid & 31;
    const int wave = tid >> 5;           // 0..15
    const int wm   = wave >> 2;          // 0..3
    const int wn   = wave & 3;           // 0..3
    const int m0   = blockIdx.y * GBM;
    const int n0   = blockIdx.x * GBN;
    const int hi   = lane >> 4;
    const int l15  = lane & 15;

    v8f z = {0.f, 0.f, 0.f, 0.f, 0.f, 0.f, 0.f, 0.f};
    v8f acc00 = z, acc01 = z, acc10 = z, acc11 = z;

    const bool interior = (m0 + GBM <= M) && (n0 + GBN <= N);

    for (int k0 = 0; k0 < K; k0 += GBK) {
        const bool ktail = (k0 + GBK > K);
        if (interior && !ktail) {
            // ---- fast staging, no guards ----
            for (int i = tid; i < GBM * GKK; i += 512) {
                int r = i >> 4, c = i & 15;
                const float* p = A + (size_t)(m0 + r) * K + k0 + 2 * c;
                lA[r][c] = f2bf_pk(p[0], p[1]);
            }
            for (int i = tid; i < GBN * GKK; i += 512) {
                int c = i >> 7, n = i & 127;
                const float* p = Bm + (size_t)(k0 + 2 * c) * N + n0 + n;
                lB[n][c] = f2bf_pk(p[0], p[N]);
            }
        } else {
            // ---- guarded staging (edges / K tail), zero-padded ----
            for (int i = tid; i < GBM * GKK; i += 512) {
                int r = i >> 4, c = i & 15;
                int gm = m0 + r, gk = k0 + 2 * c;
                float v0 = (gm < M && gk     < K) ? A[(size_t)gm * K + gk]     : 0.f;
                float v1 = (gm < M && gk + 1 < K) ? A[(size_t)gm * K + gk + 1] : 0.f;
                lA[r][c] = f2bf_pk(v0, v1);
            }
            for (int i = tid; i < GBN * GKK; i += 512) {
                int c = i >> 7, n = i & 127;
                int gk = k0 + 2 * c, gn = n0 + n;
                float v0 = (gk     < K && gn < N) ? Bm[(size_t)gk * N + gn]       : 0.f;
                float v1 = (gk + 1 < K && gn < N) ? Bm[(size_t)(gk + 1) * N + gn] : 0.f;
                lB[n][c] = f2bf_pk(v0, v1);
            }
        }
        // prefetch next K-panel (lowers to global_prefetch_b8)
        if (k0 + GBK < K) {
            int pr = m0 + (tid & 127); if (pr >= M) pr = M - 1;
            __builtin_prefetch(&A[(size_t)pr * K + (k0 + GBK)], 0, 1);
            int br = k0 + GBK + (tid & 31); if (br >= K) br = K - 1;
            __builtin_prefetch(&Bm[(size_t)br * N + n0], 0, 1);
        }
        __syncthreads();

        // ---- fragments (ISA 7.12.2 wave32 16-bit layouts, u32-pair loads) --
        BfFrag fa0, fa1, fb0, fb1;
        const int mr0 = wm * 32 + l15, mr1 = mr0 + 16;
        const int nc0 = wn * 32 + l15, nc1 = nc0 + 16;
#pragma unroll
        for (int p = 0; p < 8; ++p) {
            int ia = (p & 3) + 8 * (p >> 2) + 4 * hi;   // A 16x32: k-pair index
            fa0.q[p] = lA[mr0][ia];
            fa1.q[p] = lA[mr1][ia];
            int ib = (p & 3) + 8 * (p >> 2) + 4 * hi;   // B 32x16: k-pair index
            fb0.q[p] = lB[nc0][ib];
            fb1.q[p] = lB[nc1][ib];
        }
        acc00 = __builtin_amdgcn_wmma_f32_16x16x32_bf16(false, fa0.v, false, fb0.v, (short)0, acc00, false, false);
        acc01 = __builtin_amdgcn_wmma_f32_16x16x32_bf16(false, fa0.v, false, fb1.v, (short)0, acc01, false, false);
        acc10 = __builtin_amdgcn_wmma_f32_16x16x32_bf16(false, fa1.v, false, fb0.v, (short)0, acc10, false, false);
        acc11 = __builtin_amdgcn_wmma_f32_16x16x32_bf16(false, fa1.v, false, fb1.v, (short)0, acc11, false, false);
        __syncthreads();
    }

    // ---- epilogue: C/D layout VGPR r -> M = r + 8*(lane/16), N = lane%16 ---
#pragma unroll
    for (int sm = 0; sm < 2; ++sm) {
#pragma unroll
        for (int sn = 0; sn < 2; ++sn) {
            v8f* accp = sm == 0 ? (sn == 0 ? &acc00 : &acc01)
                                : (sn == 0 ? &acc10 : &acc11);
            int nb = n0 + wn * 32 + sn * 16 + l15;
            int mb = m0 + wm * 32 + sm * 16 + 8 * hi;
#pragma unroll
            for (int r = 0; r < 8; ++r) {
                int gm = mb + r;
                if (gm < M && nb < N) {
                    float v = (*accp)[r];
                    if (bias)  v += bias[nb];
                    if (resid) v += resid[(size_t)gm * N + nb];
                    if (act == 1)      v = v > 0.f ? v : 0.f;
                    else if (act == 2) v = v > 0.f ? v : 0.01f * v;
                    C[(size_t)gm * N + nb] = v;
                }
            }
        }
    }
}

// ---------------------------------------------------------------------------
// wave32 block reductions (256-thread blocks = 8 waves)
// ---------------------------------------------------------------------------
__device__ __forceinline__ float block_reduce_sum(float v, float* red) {
    for (int off = 16; off > 0; off >>= 1) v += __shfl_down(v, off, 32);
    const int lane = threadIdx.x & 31, w = threadIdx.x >> 5;
    __syncthreads();
    if (lane == 0) red[w] = v;
    __syncthreads();
    if (w == 0) {
        float t = (lane < 8) ? red[lane] : 0.f;
        for (int off = 4; off > 0; off >>= 1) t += __shfl_down(t, off, 32);
        if (lane == 0) red[0] = t;
    }
    __syncthreads();
    return red[0];
}

__device__ __forceinline__ float block_reduce_max(float v, float* red) {
    for (int off = 16; off > 0; off >>= 1) v = fmaxf(v, __shfl_down(v, off, 32));
    const int lane = threadIdx.x & 31, w = threadIdx.x >> 5;
    __syncthreads();
    if (lane == 0) red[w] = v;
    __syncthreads();
    if (w == 0) {
        float t = (lane < 8) ? red[lane] : -3.4e38f;
        for (int off = 4; off > 0; off >>= 1) t = fmaxf(t, __shfl_down(t, off, 32));
        if (lane == 0) red[0] = t;
    }
    __syncthreads();
    return red[0];
}

// ---------------------------------------------------------------------------
// Multi-head attention (one block per (t, h, b)); D split into H heads of hd.
// ---------------------------------------------------------------------------
__global__ __launch_bounds__(256)
void mha_kernel(const float* __restrict__ Q, const float* __restrict__ Kt,
                const float* __restrict__ V, float* __restrict__ O,
                int T, int D, int hd, float scale)
{
    __shared__ float sc[304];   // T <= 300
    __shared__ float qs[512];   // hd <= 500
    __shared__ float red[8];
    const int b = blockIdx.z, h = blockIdx.y, t = blockIdx.x;
    const size_t qoff = ((size_t)b * T + t) * D + (size_t)h * hd;

    for (int d = threadIdx.x; d < hd; d += 256) qs[d] = Q[qoff + d];
    __syncthreads();

    for (int j = threadIdx.x; j < T; j += 256) {
        const float* kr = Kt + ((size_t)b * T + j) * D + (size_t)h * hd;
        float a = 0.f;
        for (int d = 0; d < hd; ++d) a += qs[d] * kr[d];
        sc[j] = a * scale;
    }
    __syncthreads();

    float m = -3.4e38f;
    for (int j = threadIdx.x; j < T; j += 256) m = fmaxf(m, sc[j]);
    m = block_reduce_max(m, red);

    float s = 0.f;
    for (int j = threadIdx.x; j < T; j += 256) { float e = __expf(sc[j] - m); sc[j] = e; s += e; }
    s = block_reduce_sum(s, red);
    const float inv = 1.f / s;

    for (int d = threadIdx.x; d < hd; d += 256) {
        float a = 0.f;
        for (int j = 0; j < T; ++j)
            a += sc[j] * V[((size_t)b * T + j) * D + (size_t)h * hd + d];
        O[qoff + d] = a * inv;
    }
}

// ---------------------------------------------------------------------------
// LayerNorm in place, one block per row
// ---------------------------------------------------------------------------
__global__ __launch_bounds__(256)
void layernorm_kernel(float* __restrict__ X, const float* __restrict__ g,
                      const float* __restrict__ bta, int D)
{
    __shared__ float red[8];
    float* row = X + (size_t)blockIdx.x * D;
    float s = 0.f, s2 = 0.f;
    for (int d = threadIdx.x; d < D; d += 256) { float v = row[d]; s += v; s2 += v * v; }
    s  = block_reduce_sum(s, red);
    s2 = block_reduce_sum(s2, red);
    float mean = s / (float)D;
    float var  = s2 / (float)D - mean * mean;
    float rstd = rsqrtf(var + 1e-5f);
    for (int d = threadIdx.x; d < D; d += 256)
        row[d] = (row[d] - mean) * rstd * g[d] + bta[d];
}

// ---------------------------------------------------------------------------
// Softmax in place over last dim, one block per row
// ---------------------------------------------------------------------------
__global__ __launch_bounds__(256)
void softmax_kernel(float* __restrict__ X, int L)
{
    __shared__ float red[8];
    float* row = X + (size_t)blockIdx.x * L;
    float m = -3.4e38f;
    for (int i = threadIdx.x; i < L; i += 256) m = fmaxf(m, row[i]);
    m = block_reduce_max(m, red);
    float s = 0.f;
    for (int i = threadIdx.x; i < L; i += 256) { float e = __expf(row[i] - m); row[i] = e; s += e; }
    s = block_reduce_sum(s, red);
    float inv = 1.f / s;
    for (int i = threadIdx.x; i < L; i += 256) row[i] *= inv;
}

// ---------------------------------------------------------------------------
// Positional encoding add (sin/cos), elementwise
// ---------------------------------------------------------------------------
__global__ void posenc_add_kernel(float* __restrict__ X, int T, int D, int total)
{
    int i = blockIdx.x * blockDim.x + threadIdx.x;
    if (i >= total) return;
    int d = i % D;
    int t = (i / D) % T;
    int i2 = d & ~1;
    float ang = (float)t * __expf(-(float)i2 / (float)D * 9.210340371976184f);
    X[i] += (d & 1) ? __cosf(ang) : __sinf(ang);
}

// ---------------------------------------------------------------------------
// Build "all features": AF[b, c, l*5+j] = concat(L,R)[b, l, c*5+j]
// ---------------------------------------------------------------------------
__global__ void build_af_kernel(const float* __restrict__ L, const float* __restrict__ R,
                                float* __restrict__ AF, int total)
{
    int i = blockIdx.x * blockDim.x + threadIdx.x;
    if (i >= total) return;
    int mcol = i % 1500;
    int c = (i / 1500) % 60;
    int b = i / (1500 * 60);
    int l = mcol / 5, j = mcol % 5;
    int f = c * 5 + j;
    AF[i] = (f < 150) ? L[((size_t)b * 300 + l) * 150 + f]
                      : R[((size_t)b * 300 + l) * 150 + (f - 150)];
}

// ---------------------------------------------------------------------------
// Fold  left_lin -> att_w  into one vector: wcomb[0..149], wcomb[150]=bias
// ---------------------------------------------------------------------------
__global__ void wcomb_kernel(const float* __restrict__ lin_w, const float* __restrict__ lin_b,
                             const float* __restrict__ att_w, const float* __restrict__ att_b,
                             float* __restrict__ out)
{
    int d = threadIdx.x;
    if (d < 150) {
        float a = 0.f;
        for (int e = 0; e < 256; ++e) a += lin_w[d * 256 + e] * att_w[e];
        out[d] = a;
    } else if (d == 150) {
        float a = 0.f;
        for (int e = 0; e < 256; ++e) a += lin_b[e] * att_w[e];
        out[150] = a + att_b[0];
    }
}

// ---------------------------------------------------------------------------
// att[row] = leaky(dot(LE[row,:150], wcomb) + wcomb[150]); one wave per row
// ---------------------------------------------------------------------------
__global__ __launch_bounds__(256)
void att_score_kernel(const float* __restrict__ LE, const float* __restrict__ wcomb,
                      float* __restrict__ att, int rows, int D)
{
    int row = blockIdx.x * 8 + (threadIdx.x >> 5);
    int lane = threadIdx.x & 31;
    if (row >= rows) return;
    float a = 0.f;
    for (int d = lane; d < D; d += 32) a += LE[(size_t)row * D + d] * wcomb[d];
    for (int off = 16; off > 0; off >>= 1) a += __shfl_down(a, off, 32);
    if (lane == 0) {
        float v = a + wcomb[D];
        att[row] = v > 0.f ? v : 0.01f * v;
    }
}

// ---------------------------------------------------------------------------
// ctx[b,d] = sum_{a=0..4} att[b, 59*5+a] * pres[b, 59-a, d]   (only t=59 kept)
// ---------------------------------------------------------------------------
__global__ void ctx_kernel(const float* __restrict__ att, const float* __restrict__ pres,
                           float* __restrict__ ctx)
{
    int b = blockIdx.x, d = threadIdx.x;  // 256 threads
    float acc = 0.f;
    for (int a = 0; a < 5; ++a)
        acc += att[b * 300 + 59 * 5 + a] * pres[((size_t)b * 60 + (59 - a)) * 256 + d];
    ctx[b * 256 + d] = acc;
}

__global__ void copy_kernel(const float* __restrict__ s, float* __restrict__ d, int n)
{
    int i = blockIdx.x * blockDim.x + threadIdx.x;
    if (i < n) d[i] = s[i];
}

// ---------------------------------------------------------------------------
// Host-side helpers
// ---------------------------------------------------------------------------
static inline void gemm(const float* A, const float* Bm, const float* bias,
                        const float* resid, float* C, int M, int N, int K, int act,
                        hipStream_t s)
{
    dim3 g((N + GBN - 1) / GBN, (M + GBM - 1) / GBM);
    wmma_gemm_kernel<<<g, 512, 0, s>>>(A, Bm, bias, resid, C, M, N, K, act);
}

// Transformer encoder stack (posenc applied by caller). Layer params at
// d_in[base + 16*l + k]:  wq bq wk bk wv bv wo bo ln1g ln1b w1 b1 w2 b2 ln2g ln2b
static void run_encoder(float** px, float* f1, float* f2, float* f3, float* f4,
                        float* hid, int B, int T, int D, int H, int depth,
                        void* const* d_in, int base, int chunkRows, hipStream_t stream)
{
    const int M = B * T;
    const int hd = D / H;
    const float scale = 1.0f / sqrtf((float)hd);
    float* x = *px;
    for (int l = 0; l < depth; ++l) {
        const float* wq   = (const float*)d_in[base + 16 * l + 0];
        const float* bq   = (const float*)d_in[base + 16 * l + 1];
        const float* wk   = (const float*)d_in[base + 16 * l + 2];
        const float* bk   = (const float*)d_in[base + 16 * l + 3];
        const float* wv   = (const float*)d_in[base + 16 * l + 4];
        const float* bv   = (const float*)d_in[base + 16 * l + 5];
        const float* wo   = (const float*)d_in[base + 16 * l + 6];
        const float* bo   = (const float*)d_in[base + 16 * l + 7];
        const float* ln1g = (const float*)d_in[base + 16 * l + 8];
        const float* ln1b = (const float*)d_in[base + 16 * l + 9];
        const float* w1   = (const float*)d_in[base + 16 * l + 10];
        const float* b1   = (const float*)d_in[base + 16 * l + 11];
        const float* w2   = (const float*)d_in[base + 16 * l + 12];
        const float* b2   = (const float*)d_in[base + 16 * l + 13];
        const float* ln2g = (const float*)d_in[base + 16 * l + 14];
        const float* ln2b = (const float*)d_in[base + 16 * l + 15];

        gemm(x, wq, bq, nullptr, f1, M, D, D, 0, stream);                 // Q
        gemm(x, wk, bk, nullptr, f2, M, D, D, 0, stream);                 // K
        gemm(x, wv, bv, nullptr, f3, M, D, D, 0, stream);                 // V
        mha_kernel<<<dim3(T, H, B), 256, 0, stream>>>(f1, f2, f3, f4, T, D, hd, scale);
        gemm(f4, wo, bo, x, f2, M, D, D, 0, stream);                      // O@wo + x
        layernorm_kernel<<<M, 256, 0, stream>>>(f2, ln1g, ln1b, D);       // LN1 -> f2
        for (int r0 = 0; r0 < M; r0 += chunkRows) {
            int cm = (M - r0 < chunkRows) ? (M - r0) : chunkRows;
            gemm(f2 + (size_t)r0 * D, w1, b1, nullptr, hid, cm, 1024, D, 1, stream);
            gemm(hid, w2, b2, f2 + (size_t)r0 * D, f3 + (size_t)r0 * D, cm, D, 1024, 0, stream);
        }
        layernorm_kernel<<<M, 256, 0, stream>>>(f3, ln2g, ln2b, D);       // LN2 -> f3
        float* t = x; x = f3; f3 = t;                                     // x <- f3
    }
    *px = x;
}

// ---------------------------------------------------------------------------
// Entry point
// ---------------------------------------------------------------------------
extern "C" void kernel_launch(void* const* d_in, const int* in_sizes, int n_in,
                              void* d_out, int out_size, void* d_ws, size_t ws_size,
                              hipStream_t stream)
{
    (void)in_sizes; (void)n_in; (void)out_size; (void)ws_size;
    const float* leftF  = (const float*)d_in[1];
    const float* rightF = (const float*)d_in[2];
    float* out = (float*)d_out;
    float* ws  = (float*)d_ws;

    // workspace layout (floats); largest activation = 128*60*1500 = 11,520,000
    const size_t BUF = 11520000;
    float* s0   = ws;
    float* s1   = ws + 1 * BUF;
    float* s2   = ws + 2 * BUF;
    float* s3   = ws + 3 * BUF;
    float* s4   = ws + 4 * BUF;
    float* hid  = ws + 5 * BUF;          // 7680*1024 = 7,864,320
    float* att  = hid + 7864320;         // 128*300   = 38,400
    float* pres = att + 38400;           // 128*60*256 = 1,966,080
    float* ctx  = pres + 1966080;        // 128*256   = 32,768
    float* h1   = ctx + 32768;           // 128*128
    float* h2   = h1 + 16384;            // 128*64
    float* h3   = h2 + 8192;             // 128*32
    float* wcmb = h3 + 4096;             // 151

    // ---------------- left encoder (right encoder is dead code) ------------
    hipMemcpyAsync(s0, leftF, (size_t)128 * 300 * 150 * sizeof(float),
                   hipMemcpyDeviceToDevice, stream);
    {
        int total = 128 * 300 * 150;
        posenc_add_kernel<<<(total + 255) / 256, 256, 0, stream>>>(s0, 300, 150, total);
    }
    float* xL = s0;
    run_encoder(&xL, s1, s2, s3, s4, hid, 128, 300, 150, 3, 2, d_in, /*base=*/3,
                /*chunk=*/7680, stream);

    // ---------------- attention weights over left encoding -----------------
    wcomb_kernel<<<1, 256, 0, stream>>>((const float*)d_in[99], (const float*)d_in[100],
                                        (const float*)d_in[105], (const float*)d_in[106], wcmb);
    att_score_kernel<<<(38400 + 7) / 8, 256, 0, stream>>>(xL, wcmb, att, 38400, 150);
    softmax_kernel<<<128, 256, 0, stream>>>(att, 300);
    copy_kernel<<<(38400 + 255) / 256, 256, 0, stream>>>(att, out + 1152, 38400);

    // ---------------- "all" encoder ----------------------------------------
    {
        int total = 128 * 60 * 1500;
        build_af_kernel<<<(total + 255) / 256, 256, 0, stream>>>(leftF, rightF, s0, total);
        posenc_add_kernel<<<(total + 255) / 256, 256, 0, stream>>>(s0, 60, 1500, total);
    }
    float* xA = s0;
    run_encoder(&xA, s1, s2, s3, s4, hid, 128, 60, 1500, 3, 2, d_in, /*base=*/67,
                /*chunk=*/7680, stream);

    // pres = softmax(xA @ enc_all_w + enc_all_b)
    gemm(xA, (const float*)d_in[103], (const float*)d_in[104], nullptr, pres,
         7680, 256, 1500, 0, stream);
    softmax_kernel<<<7680, 256, 0, stream>>>(pres, 256);

    // ctx at t = 59 only (pred keeps only the last time step)
    ctx_kernel<<<128, 256, 0, stream>>>(att, pres, ctx);

    // MLP head (M = 128)
    gemm(ctx, (const float*)d_in[107], (const float*)d_in[108], nullptr, h1, 128, 128, 256, 2, stream);
    gemm(h1,  (const float*)d_in[109], (const float*)d_in[110], nullptr, h2, 128,  64, 128, 2, stream);
    gemm(h2,  (const float*)d_in[111], (const float*)d_in[112], nullptr, h3, 128,  32,  64, 2, stream);
    gemm(h3,  (const float*)d_in[113], (const float*)d_in[114], nullptr, out, 128,  9,  32, 0, stream);
}
